// PointTransformerBlock_4054449128184
// MI455X (gfx1250) — matrix operand order repaired and checked by reading, not compiled, not run on hardware
//
#include <hip/hip_runtime.h>
#include <hip/hip_bf16.h>

// ---------------------------------------------------------------------------
// Point Transformer block, fused for MI455X (gfx1250, wave32, WMMA).
// Dominant cost: two 128x128 MLPs over B*N*K = 262144 rows (~17.2 GFLOP) plus
// the k/v gathers (~0.5 GB fused). All GEMMs use v_wmma_f32_16x16x32_f16
// (fp16 in, fp32 accumulate); gathers/stores vectorized to b128.
// ---------------------------------------------------------------------------

typedef __attribute__((ext_vector_type(16))) _Float16 v16h;
typedef __attribute__((ext_vector_type(8)))  _Float16 v8h;
typedef __attribute__((ext_vector_type(4)))  _Float16 v4h;
typedef __attribute__((ext_vector_type(8)))  float    v8f;

#define NPTS 4096
#define CCH  128
#define KNB  32     // neighbors per point
#define PPB  2      // points per attention workgroup
#define AS   136    // LDS row stride in halves (128 + 8 pad)
#define OS   66     // k_final LDS out-tile stride in floats (64 + 2 pad)

struct PT {
  // inputs (fp32)
  const float *xyz, *feat;
  const float *W1, *Wq, *bq, *Wk, *bk, *Wv, *bv;
  const float *Wp1, *Wp2, *bp2;
  const float *Wa1, *Wa2, *ba2, *W2;
  const float *bn1[4], *bnp[4], *bna1[4], *bna2[4], *bnt[4], *bn2[4];
  // workspace
  _Float16 *w16;    // 7 x 16384 fp16 weights: W1,Wq,Wk,Wv,Wa1,Wa2,W2
  float    *cs;     // BN consts: [bn1 s|h][bna1][bna2][bnt][bn2] @256 each, bnp @1280
  _Float16 *y16;    // 8192x128 (post lin1+bn1+relu)
  float    *q, *k, *v;  // 8192x128 each
  _Float16 *yatt;   // 8192x128 (post attention + bn_t + relu)
  int      *idx;    // 8192x32
  float    *outp;   // d_out: [24576 p][B*C*N y]
};

// --- WMMA fragment loaders (contiguous K order k = 16*(lane/16) + s in A and B, wave32) ----------
__device__ inline v16h load_a_frag(const _Float16* sA, int mr, int kb, int lane) {
  int M  = lane & 15;
  int k0 = kb + ((lane >> 4) << 4);          // half-wave K split: +0 / +16
  const _Float16* p = sA + (mr + M) * AS + k0;
  v8h lo = *(const v8h*)(p);                 // K = k0 .. k0+7
  v8h hi = *(const v8h*)(p + 8);             // K = k0+8 .. k0+15
  v16h a;
#pragma unroll
  for (int i = 0; i < 8; ++i) { a[i] = lo[i]; a[8 + i] = hi[i]; }
  return a;
}

// B[k][n] = W[n][k]; 16 consecutive K halves per lane -> one 32B load.
__device__ inline v16h load_b_frag(const _Float16* W, int nb, int kb, int lane) {
  int ch = nb + (lane & 15);
  int k0 = kb + ((lane >> 4) << 4);          // half-wave K split: +0 / +16
  return *(const v16h*)(W + ch * CCH + k0);
}

// 8-wave GEMM: 64x128 output block, wave -> (mtile = w>>1, n-half = w&1)
__device__ inline void gemm_core_w8(const _Float16* sA, const _Float16* W,
                                    v8f acc[4], int wave, int lane) {
  int mr  = (wave >> 1) << 4;
  int nb0 = (wave & 1) << 6;
#pragma unroll
  for (int kk = 0; kk < 4; ++kk) {
    v16h af = load_a_frag(sA, mr, kk << 5, lane);
#pragma unroll
    for (int t = 0; t < 4; ++t) {
      v16h bf = load_b_frag(W, nb0 + (t << 4), kk << 5, lane);
      acc[t] = __builtin_amdgcn_wmma_f32_16x16x32_f16(
          false, af, false, bf, (short)0, acc[t], false, false);
    }
  }
}

// 4-wave GEMM: 64x128 block, wave owns one 16-row stripe, all 8 n-tiles
__device__ inline void gemm_rows64_full(const _Float16* sA, const _Float16* W,
                                        v8f acc[8], int wave, int lane) {
  int mr = wave << 4;
#pragma unroll
  for (int kk = 0; kk < 4; ++kk) {
    v16h af = load_a_frag(sA, mr, kk << 5, lane);
#pragma unroll
    for (int t = 0; t < 8; ++t) {
      v16h bf = load_b_frag(W, t << 4, kk << 5, lane);
      acc[t] = __builtin_amdgcn_wmma_f32_16x16x32_f16(
          false, af, false, bf, (short)0, acc[t], false, false);
    }
  }
}

// --- prep: fp16 weights, BN scale/shift, copy p output ---------------------
__global__ __launch_bounds__(256) void k_prep(PT pt) {
  int i = blockIdx.x * 256 + threadIdx.x;    // grid covers 7*16384
  if (i < 7 * 16384) {
    const float* srcs[7] = {pt.W1, pt.Wq, pt.Wk, pt.Wv, pt.Wa1, pt.Wa2, pt.W2};
    pt.w16[i] = (_Float16)srcs[i >> 14][i & 16383];
  }
  if (i < 5 * 128) {
    int bnid = i >> 7, c = i & 127;
    const float* const* p = (bnid == 0) ? pt.bn1 : (bnid == 1) ? pt.bna1
                          : (bnid == 2) ? pt.bna2 : (bnid == 3) ? pt.bnt : pt.bn2;
    float s = p[0][c] * rsqrtf(p[3][c] + 1e-5f);
    pt.cs[bnid * 256 + c]       = s;
    pt.cs[bnid * 256 + 128 + c] = p[1][c] - p[2][c] * s;
  }
  if (i < 3) {
    float s = pt.bnp[0][i] * rsqrtf(pt.bnp[3][i] + 1e-5f);
    pt.cs[1280 + i] = s;
    pt.cs[1283 + i] = pt.bnp[1][i] - pt.bnp[2][i] * s;
  }
  if (i < 2 * NPTS * 3) pt.outp[i] = pt.xyz[i];   // p output (tuple elem 0)
}

// --- KNN: LDS-tiled squared distances, register top-32 ---------------------
__global__ __launch_bounds__(256) void k_knn(PT pt) {
  int gp = blockIdx.x * 256 + threadIdx.x;   // 0..8191
  int b  = gp >> 12;
  int n  = gp & (NPTS - 1);
  const float* P = pt.xyz + (long)b * NPTS * 3;
  float px = P[n * 3], py = P[n * 3 + 1], pz = P[n * 3 + 2];
  float bd[KNB]; int bi[KNB];
#pragma unroll
  for (int t = 0; t < KNB; ++t) { bd[t] = 3.0e38f; bi[t] = 0; }
  float wmax = 3.0e38f; int wslot = 0;
  __shared__ float sx[256], sy[256], sz[256];
  for (int j0 = 0; j0 < NPTS; j0 += 256) {
    __syncthreads();
    int j = j0 + threadIdx.x;
    sx[threadIdx.x] = P[j * 3];
    sy[threadIdx.x] = P[j * 3 + 1];
    sz[threadIdx.x] = P[j * 3 + 2];
    __syncthreads();
    for (int jj = 0; jj < 256; ++jj) {
      float dx = sx[jj] - px, dy = sy[jj] - py, dz = sz[jj] - pz;
      float d = dx * dx + dy * dy + dz * dz;
      if (d < wmax) {
        bd[wslot] = d; bi[wslot] = j0 + jj;
        wmax = bd[0]; wslot = 0;
#pragma unroll
        for (int t = 1; t < KNB; ++t)
          if (bd[t] > wmax) { wmax = bd[t]; wslot = t; }
      }
    }
  }
#pragma unroll
  for (int t = 0; t < KNB; ++t) pt.idx[(long)gp * KNB + t] = bi[t];
}

// --- lin1: y = relu(bn1(x @ W1^T)), x transposed from features -------------
__global__ __launch_bounds__(256) void k_lin1(PT pt) {
  __shared__ _Float16 sA[64 * AS];
  int tid = threadIdx.x, lane = tid & 31, wave = tid >> 5;
  long gr0 = (long)blockIdx.x * 64;
  int b = (int)(gr0 >> 12), n0 = (int)(gr0 & (NPTS - 1));
  const float* F = pt.feat + (long)b * CCH * NPTS;
  // float4 along n: coalesced b128 loads of the [C,N] source
  for (int i = tid; i < 64 * CCH / 4; i += 256) {
    int c  = i >> 4;
    int rr = (i & 15) << 2;
    float4 f = *(const float4*)(F + (long)c * NPTS + n0 + rr);
    sA[(rr + 0) * AS + c] = (_Float16)f.x;
    sA[(rr + 1) * AS + c] = (_Float16)f.y;
    sA[(rr + 2) * AS + c] = (_Float16)f.z;
    sA[(rr + 3) * AS + c] = (_Float16)f.w;
  }
  __syncthreads();
  v8f acc[4];
#pragma unroll
  for (int t = 0; t < 4; ++t) { v8f z = {}; acc[t] = z; }
  gemm_core_w8(sA, pt.w16, acc, wave, lane);
  int mr = (wave >> 1) << 4, nb0 = (wave & 1) << 6;
#pragma unroll
  for (int t = 0; t < 4; ++t)
#pragma unroll
    for (int j = 0; j < 8; ++j) {
      int row = mr + j + ((lane >> 4) << 3);
      int col = nb0 + (t << 4) + (lane & 15);
      float yv = fmaxf(acc[t][j] * pt.cs[col] + pt.cs[128 + col], 0.f);
      pt.y16[(gr0 + row) * CCH + col] = (_Float16)yv;
    }
}

// --- q/k/v projections: gridDim.z selects weight/bias/output ---------------
__global__ __launch_bounds__(256) void k_qkv(PT pt) {
  int z = blockIdx.z;
  const _Float16* W = pt.w16 + (size_t)(1 + z) * 16384;
  const float* bias = (z == 0) ? pt.bq : (z == 1) ? pt.bk : pt.bv;
  float* dst        = (z == 0) ? pt.q  : (z == 1) ? pt.k  : pt.v;
  __shared__ _Float16 sA[64 * AS];
  int tid = threadIdx.x, lane = tid & 31, wave = tid >> 5;
  long gr0 = (long)blockIdx.x * 64;
  for (int i = tid; i < 64 * CCH / 8; i += 256) {   // 16B chunks
    int rr = i >> 4;
    int c8 = (i & 15) << 3;
    *(v8h*)(sA + rr * AS + c8) = *(const v8h*)(pt.y16 + (gr0 + rr) * CCH + c8);
  }
  __syncthreads();
  v8f acc[4];
#pragma unroll
  for (int t = 0; t < 4; ++t) { v8f zz = {}; acc[t] = zz; }
  gemm_core_w8(sA, W, acc, wave, lane);
  int mr = (wave >> 1) << 4, nb0 = (wave & 1) << 6;
#pragma unroll
  for (int t = 0; t < 4; ++t)
#pragma unroll
    for (int j = 0; j < 8; ++j) {
      int row = mr + j + ((lane >> 4) << 3);
      int col = nb0 + (t << 4) + (lane & 15);
      dst[(gr0 + row) * CCH + col] = acc[t][j] + bias[col];
    }
}

// --- fused attention: 2 points/WG, gathers + posMLP + 2 GEMMs + softmax ----
__global__ __launch_bounds__(128) void k_attn(PT pt) {
  __shared__ _Float16 sA[64 * AS];    // a1 (GEMM1 in), then a3 (softmax in)
  __shared__ _Float16 sB[64 * AS];    // a2 (GEMM2 in)
  __shared__ _Float16 sNV[64 * AS];   // v + r
  __shared__ float sT[64][4];
  __shared__ float sQ[PPB * CCH];
  __shared__ float sWp2[CCH * 3];
  __shared__ float sBp2[CCH], sSc1[CCH], sSh1[CCH];
  __shared__ int   sIdx[64];
  __shared__ float sP[8];

  int tid = threadIdx.x, lane = tid & 31, wave = tid >> 5;
  long gp0 = (long)blockIdx.x * PPB;
  int b = (int)(gp0 >> 12);

  if (tid < 64) sIdx[tid] = pt.idx[gp0 * KNB + tid];
  if (tid < 6)  sP[tid]   = pt.xyz[gp0 * 3 + tid];
  sQ[tid]       = pt.q[gp0 * CCH + tid];
  sQ[tid + 128] = pt.q[gp0 * CCH + 128 + tid];
  for (int i = tid; i < CCH * 3; i += 128) sWp2[i] = pt.Wp2[i];
  sBp2[tid] = pt.bp2[tid];
  sSc1[tid] = pt.cs[256 + tid];
  sSh1[tid] = pt.cs[384 + tid];
  __syncthreads();

  if (tid < 64) {                           // positional MLP hidden (3-vec)
    int p = tid >> 5;
    const float* Pn = pt.xyz + ((long)b * NPTS + sIdx[tid]) * 3;
    float d0 = Pn[0] - sP[p * 3], d1 = Pn[1] - sP[p * 3 + 1], d2 = Pn[2] - sP[p * 3 + 2];
#pragma unroll
    for (int i = 0; i < 3; ++i) {
      float u = pt.Wp1[i * 3] * d0 + pt.Wp1[i * 3 + 1] * d1 + pt.Wp1[i * 3 + 2] * d2;
      sT[tid][i] = fmaxf(u * pt.cs[1280 + i] + pt.cs[1283 + i], 0.f);
    }
  }
  __syncthreads();

  { // build a1 = relu(bna1(q - k[idx] + r)) and nv = v[idx] + r
    // float4 gathers: each gathered row is 512B contiguous -> b128 loads
    for (int i = tid; i < 64 * (CCH / 4); i += 128) {
      int row = i >> 5;
      int c   = (i & 31) << 2;
      int p   = row >> 5;
      long base = ((long)b * NPTS + sIdx[row]) * CCH + c;
      float4 kv = *(const float4*)(pt.k + base);
      float4 vv = *(const float4*)(pt.v + base);
      float t0 = sT[row][0], t1 = sT[row][1], t2 = sT[row][2];
      v4h a4, n4;
#pragma unroll
      for (int u = 0; u < 4; ++u) {
        int cc = c + u;
        float kx = (&kv.x)[u], vx = (&vv.x)[u];
        float r  = sWp2[cc * 3] * t0 + sWp2[cc * 3 + 1] * t1 + sWp2[cc * 3 + 2] * t2 + sBp2[cc];
        float ar = sQ[p * CCH + cc] - kx + r;
        a4[u] = (_Float16)fmaxf(ar * sSc1[cc] + sSh1[cc], 0.f);
        n4[u] = (_Float16)(vx + r);
      }
      *(v4h*)(sA  + row * AS + c) = a4;
      *(v4h*)(sNV + row * AS + c) = n4;
    }
  }
  __syncthreads();

  v8f acc[8];
#pragma unroll
  for (int t = 0; t < 8; ++t) { v8f z = {}; acc[t] = z; }
  gemm_rows64_full(sA, pt.w16 + (size_t)4 * 16384, acc, wave, lane);  // Wa1
#pragma unroll
  for (int t = 0; t < 8; ++t)                 // a2 = relu(bna2(h)) -> sB
#pragma unroll
    for (int j = 0; j < 8; ++j) {
      int row = (wave << 4) + j + ((lane >> 4) << 3);
      int col = (t << 4) + (lane & 15);
      float h = fmaxf(acc[t][j] * pt.cs[512 + col] + pt.cs[640 + col], 0.f);
      sB[row * AS + col] = (_Float16)h;
    }
  __syncthreads();

#pragma unroll
  for (int t = 0; t < 8; ++t) { v8f z = {}; acc[t] = z; }
  gemm_rows64_full(sB, pt.w16 + (size_t)5 * 16384, acc, wave, lane);  // Wa2
#pragma unroll
  for (int t = 0; t < 8; ++t)                 // a3 = h + ba2 -> sA (reuse)
#pragma unroll
    for (int j = 0; j < 8; ++j) {
      int row = (wave << 4) + j + ((lane >> 4) << 3);
      int col = (t << 4) + (lane & 15);
      sA[row * AS + col] = (_Float16)(acc[t][j] + pt.ba2[col]);
    }
  __syncthreads();

  { // softmax over K per (point, channel) + weighted sum + bn_t + relu
    int c = tid;
#pragma unroll
    for (int p = 0; p < PPB; ++p) {
      const _Float16* col = sA  + (p * KNB) * AS + c;
      const _Float16* nvc = sNV + (p * KNB) * AS + c;
      float mx = -3.0e38f;
#pragma unroll
      for (int k2 = 0; k2 < KNB; ++k2) mx = fmaxf(mx, (float)col[k2 * AS]);
      float sum = 0.f, av = 0.f;
#pragma unroll
      for (int k2 = 0; k2 < KNB; ++k2) {
        float e = __expf((float)col[k2 * AS] - mx);
        sum += e;
        av  += e * (float)nvc[k2 * AS];
      }
      float y = av / sum;
      y = fmaxf(y * pt.cs[768 + c] + pt.cs[896 + c], 0.f);
      pt.yatt[(gp0 + p) * CCH + c] = (_Float16)y;
    }
  }
}

// --- final: out = relu(bn2(yatt @ W2^T) + x), LDS-transposed coalesced store
__global__ __launch_bounds__(256) void k_final(PT pt) {
  __shared__ _Float16 sA[64 * AS];
  __shared__ float sO[CCH * OS];      // [channel][row] transposed tile
  int tid = threadIdx.x, lane = tid & 31, wave = tid >> 5;
  long gr0 = (long)blockIdx.x * 64;
  int b = (int)(gr0 >> 12), n0 = (int)(gr0 & (NPTS - 1));
  for (int i = tid; i < 64 * CCH / 8; i += 256) {
    int rr = i >> 4;
    int c8 = (i & 15) << 3;
    *(v8h*)(sA + rr * AS + c8) = *(const v8h*)(pt.yatt + (gr0 + rr) * CCH + c8);
  }
  __syncthreads();
  v8f acc[4];
#pragma unroll
  for (int t = 0; t < 4; ++t) { v8f z = {}; acc[t] = z; }
  gemm_core_w8(sA, pt.w16 + (size_t)6 * 16384, acc, wave, lane);
  int mr = (wave >> 1) << 4, nb0 = (wave & 1) << 6;
#pragma unroll
  for (int t = 0; t < 4; ++t)
#pragma unroll
    for (int j = 0; j < 8; ++j) {
      int row = mr + j + ((lane >> 4) << 3);
      int col = nb0 + (t << 4) + (lane & 15);
      sO[col * OS + row] = acc[t][j] * pt.cs[1024 + col] + pt.cs[1152 + col];
    }
  __syncthreads();
  // residual + relu + store, contiguous along n: b128 loads and stores
  const float* F = pt.feat + (long)b * CCH * NPTS;
  float* O = pt.outp + 2 * NPTS * 3 + (long)b * CCH * NPTS;
  for (int i = tid; i < 64 * CCH / 4; i += 256) {
    int c  = i >> 4;
    int rr = (i & 15) << 2;
    float4 x = *(const float4*)(F + (long)c * NPTS + n0 + rr);
    float4 o;
    o.x = fmaxf(sO[c * OS + rr + 0] + x.x, 0.f);
    o.y = fmaxf(sO[c * OS + rr + 1] + x.y, 0.f);
    o.z = fmaxf(sO[c * OS + rr + 2] + x.z, 0.f);
    o.w = fmaxf(sO[c * OS + rr + 3] + x.w, 0.f);
    *(float4*)(O + (long)c * NPTS + n0 + rr) = o;
  }
}

extern "C" void kernel_launch(void* const* d_in, const int* in_sizes, int n_in,
                              void* d_out, int out_size, void* d_ws, size_t ws_size,
                              hipStream_t stream) {
  (void)in_sizes; (void)n_in; (void)out_size; (void)ws_size;
  PT pt;
  auto F = [&](int i) { return (const float*)d_in[i]; };
  pt.xyz = F(0); pt.feat = F(1);
  pt.W1 = F(2);
  for (int j = 0; j < 4; ++j) pt.bn1[j] = F(3 + j);
  pt.Wq = F(7);  pt.bq = F(8);
  pt.Wk = F(9);  pt.bk = F(10);
  pt.Wv = F(11); pt.bv = F(12);
  pt.Wp1 = F(13);
  for (int j = 0; j < 4; ++j) pt.bnp[j] = F(14 + j);
  pt.Wp2 = F(18); pt.bp2 = F(19);
  for (int j = 0; j < 4; ++j) pt.bna1[j] = F(20 + j);
  pt.Wa1 = F(24);
  for (int j = 0; j < 4; ++j) pt.bna2[j] = F(25 + j);
  pt.Wa2 = F(29); pt.ba2 = F(30);
  for (int j = 0; j < 4; ++j) pt.bnt[j] = F(31 + j);
  pt.W2 = F(35);
  for (int j = 0; j < 4; ++j) pt.bn2[j] = F(36 + j);

  char* w = (char*)d_ws;
  size_t o = 0;
  auto alloc = [&](size_t bytes) -> void* {
    void* P = w + o;
    o += (bytes + 255) & ~(size_t)255;
    return P;
  };
  pt.w16  = (_Float16*)alloc((size_t)7 * 16384 * sizeof(_Float16));
  pt.cs   = (float*)alloc(1286 * sizeof(float));
  pt.y16  = (_Float16*)alloc((size_t)8192 * 128 * sizeof(_Float16));
  pt.q    = (float*)alloc((size_t)8192 * 128 * sizeof(float));
  pt.k    = (float*)alloc((size_t)8192 * 128 * sizeof(float));
  pt.v    = (float*)alloc((size_t)8192 * 128 * sizeof(float));
  pt.yatt = (_Float16*)alloc((size_t)8192 * 128 * sizeof(_Float16));
  pt.idx  = (int*)alloc((size_t)8192 * 32 * sizeof(int));
  pt.outp = (float*)d_out;

  k_prep <<<dim3(448),        dim3(256), 0, stream>>>(pt);
  k_knn  <<<dim3(32),         dim3(256), 0, stream>>>(pt);
  k_lin1 <<<dim3(128),        dim3(256), 0, stream>>>(pt);
  k_qkv  <<<dim3(128, 1, 3),  dim3(256), 0, stream>>>(pt);
  k_attn <<<dim3(4096),       dim3(128), 0, stream>>>(pt);
  k_final<<<dim3(128),        dim3(256), 0, stream>>>(pt);
}
